// MultiHeadAttention_34007551049751
// MI455X (gfx1250) — matrix operand order repaired and checked by reading, not compiled, not run on hardware
//
#include <hip/hip_runtime.h>
#include <hip/hip_bf16.h>
#include <math.h>

// ---------------- problem constants ----------------
#define B_   4
#define T_   2048
#define D_   1024
#define H_   16
#define HD_  64            // head dim (KD == VD)
#define BT_  (B_ * T_)     // 8192 rows

typedef __bf16 bhalf;
typedef __attribute__((ext_vector_type(16))) __bf16 v16bf;
typedef __attribute__((ext_vector_type(8)))  __bf16 v8bf;
typedef __attribute__((ext_vector_type(8)))  float  v8f;
typedef __attribute__((ext_vector_type(4)))  unsigned int u32x4;
typedef __attribute__((ext_vector_type(8)))  int i32x8;
typedef __attribute__((ext_vector_type(4)))  int i32x4;

union ABfrag { v16bf v; v8bf h[2]; };

__device__ __forceinline__ v8f wmma_bf16(v16bf a, v16bf b, v8f c) {
  // D = A(16x32) * B(32x16) + C, f32 accumulate
  return __builtin_amdgcn_wmma_f32_16x16x32_bf16(
      false, a, false, b, (short)0, c, false, false);
}

// ---------------------------------------------------------------------------
// Tensor Data Mover: issue a 2D tile load global->LDS (D# groups per ISA
// CDNA5 ISA). data_size=2B. Optional LDS row padding so fragment
// reads are bank-conflict-free. Groups 2/3 zero (2D tensor). Wave-uniform.
// This toolchain exposes the 6-arg builtin form (extra i32x8 before cpol).
// ---------------------------------------------------------------------------
__device__ __forceinline__ void tdm_load_2d(unsigned lds_addr, const void* gptr,
                                            unsigned tile_d0, unsigned tile_d1,
                                            unsigned tensor_d0, unsigned tensor_d1,
                                            unsigned stride_elems,
                                            unsigned pad_interval, unsigned pad_amount,
                                            unsigned pad_en)
{
  unsigned long long ga = (unsigned long long)gptr;
  u32x4 g0 = {
      1u,                                   // count=1 (valid descriptor)
      lds_addr,                             // bits 63:32  lds_addr (bytes)
      (unsigned)ga,                         // bits 95:64  global_addr[31:0]
      (unsigned)((ga >> 32) & 0x1FFFFFFull) | (2u << 30)  // ga[56:32] | type=2
  };
  i32x8 g1 = {
      (int)((1u << 16) | (pad_en << 20) | (pad_interval << 22) | (pad_amount << 25)),
      (int)(tensor_d0 << 16),               // tensor_dim0[15:0] @ bits 63:48
      (int)((tensor_d0 >> 16) | (tensor_d1 << 16)),
      (int)((tensor_d1 >> 16) | (tile_d0 << 16)),
      (int)tile_d1,                         // tile_dim1; tile_dim2=0
      (int)stride_elems,                    // tensor_dim0_stride[31:0]
      0, 0
  };
  i32x4 z4 = {0, 0, 0, 0};
  i32x8 z8 = {0, 0, 0, 0, 0, 0, 0, 0};
  __builtin_amdgcn_tensor_load_to_lds(g0, g1, z4, z4, z8, 0);
}

__device__ __forceinline__ unsigned lds_off(const void* p) {
  // flat LDS address: addr[31:0] is the LDS offset (ISA 10.2 aperture mapping)
  return (unsigned)(unsigned long long)p;
}

// ---------------------------------------------------------------------------
// Weight convert+transpose: WT[n][k] = bf16(W[k][n]),  1024x1024
// ---------------------------------------------------------------------------
__global__ __launch_bounds__(256)
void wconv_kernel(const float* __restrict__ W, bhalf* __restrict__ WT) {
  int gid = blockIdx.x * 256 + threadIdx.x;
  int k = gid & (D_ - 1);
  int n = gid >> 10;
  WT[(size_t)n * D_ + k] = (bhalf)W[(size_t)k * D_ + n];
}

// ---------------------------------------------------------------------------
// Tiled WMMA GEMM: out[M=8192][N=1024] = A[M][1024] @ WT^T + bias
// Block tile 128x128, 8 waves (wave tile 64x32), k-chunks of 32 via LDS.
// B tile (bf16 weights) staged by the Tensor Data Mover (wave 0 issues the
// descriptor, TDM pads rows to 80B so fragment reads are conflict-free);
// A tile staged manually (needs fp32->bf16 convert except mode 3).
// mode 0: A=f32, store bf16 (Q, scaled by 1/8 after bias)
// mode 1: A=f32, store bf16 (K)
// mode 2: A=f32, store bf16 transposed per head -> Vt[b][h][vd][t]
// mode 3: A=bf16 (ctx), store f32 to final output
// ---------------------------------------------------------------------------
#define LSTR 40  // LDS row stride in bf16 elems (80B = 64B data + 16B TDM pad)

__global__ __launch_bounds__(256)
void gemm_kernel(const float* __restrict__ Af, const bhalf* __restrict__ Ab,
                 const bhalf* __restrict__ Wt, const float* __restrict__ bias,
                 float scale, int mode,
                 bhalf* __restrict__ outb, bhalf* __restrict__ vt,
                 float* __restrict__ outf)
{
  __shared__ bhalf lA[128 * LSTR];
  __shared__ bhalf lB[128 * LSTR];

  const int tid  = threadIdx.x;
  const int lane = tid & 31;
  const int wave = tid >> 5;
  const int l16  = lane & 15;
  const int lh   = lane >> 4;          // lane half (0/1)
  const int mb = blockIdx.y * 128;
  const int nb = blockIdx.x * 128;
  const int wm = (wave & 1) * 64;      // wave M offset inside block tile
  const int wn = (wave >> 1) * 32;     // wave N offset inside block tile

  v8f acc[4][2];
#pragma unroll
  for (int ms = 0; ms < 4; ++ms)
#pragma unroll
    for (int ns = 0; ns < 2; ++ns) acc[ms][ns] = (v8f){};

  for (int kc = 0; kc < D_; kc += 32) {
    __syncthreads();
    // ---- B tile (128 n-rows x 32 k) via TDM: row 32*2B=16 dwords, pad 4 dwords
    if (wave == 0)
      tdm_load_2d(lds_off(lB), Wt + (size_t)nb * D_ + kc,
                  /*tile_d0=*/32, /*tile_d1=*/128,
                  /*tensor_d0=*/32, /*tensor_d1=*/128,
                  /*stride=*/D_, /*pad_interval(16dw)=*/3, /*pad_amount(4dw)=*/3, 1);
    // ---- stage A tile (128 x 32) ----
    if (mode == 3) {
#pragma unroll
      for (int p = 0; p < 2; ++p) {
        int r = (tid >> 2) + p * 64;
        int c = (tid & 3) * 8;
        *(v8bf*)&lA[r * LSTR + c] =
            *(const v8bf*)&Ab[(size_t)(mb + r) * D_ + kc + c];
      }
    } else {
#pragma unroll
      for (int p = 0; p < 4; ++p) {
        int r = (tid >> 3) + p * 32;
        int c = (tid & 7) * 4;
        const float4 f = *(const float4*)&Af[(size_t)(mb + r) * D_ + kc + c];
        bhalf* d = &lA[r * LSTR + c];
        d[0] = (bhalf)f.x; d[1] = (bhalf)f.y;
        d[2] = (bhalf)f.z; d[3] = (bhalf)f.w;
      }
    }
    if (wave == 0) __builtin_amdgcn_s_wait_tensorcnt(0);
    __syncthreads();

    // ---- A fragments: 16x32, lane holds row m=l16, k chunks {kb, kb+16}, kb=8*lh
    ABfrag afr[4];
#pragma unroll
    for (int ms = 0; ms < 4; ++ms) {
      int r  = wm + ms * 16 + l16;
      int kb = lh * 8;
      afr[ms].h[0] = *(v8bf*)&lA[r * LSTR + kb];
      afr[ms].h[1] = *(v8bf*)&lA[r * LSTR + kb + 16];
    }
    // ---- B fragments: lane holds col n=l16, k chunks {kb, kb+16}, kb=8*lh
    ABfrag bfr[2];
#pragma unroll
    for (int ns = 0; ns < 2; ++ns) {
      int r = wn + ns * 16 + l16;
      bfr[ns].h[0] = *(v8bf*)&lB[r * LSTR + lh * 8];
      bfr[ns].h[1] = *(v8bf*)&lB[r * LSTR + lh * 8 + 16];
    }
#pragma unroll
    for (int ms = 0; ms < 4; ++ms)
#pragma unroll
      for (int ns = 0; ns < 2; ++ns)
        acc[ms][ns] = wmma_bf16(afr[ms].v, bfr[ns].v, acc[ms][ns]);
  }

  // ---- epilogue: D value i lives at (m = i + 8*lh, n = l16) within subtile
#pragma unroll
  for (int ms = 0; ms < 4; ++ms) {
#pragma unroll
    for (int ns = 0; ns < 2; ++ns) {
      int n  = nb + wn + ns * 16 + l16;
      int m0 = mb + wm + ms * 16 + lh * 8;
      float bv = bias[n];
      if (mode == 2) {
        // V transposed per head: Vt[b][h][vd][t], 8 consecutive t -> 16B store
        int bb = m0 / T_, t0 = m0 & (T_ - 1);
        int hh = n >> 6, vd = n & 63;
        v8bf pk;
#pragma unroll
        for (int i = 0; i < 8; ++i)
          pk[i] = (bhalf)((acc[ms][ns][i] + bv) * scale);
        *(v8bf*)&vt[(((size_t)bb * H_ + hh) * HD_ + vd) * T_ + t0] = pk;
      } else if (mode == 3) {
#pragma unroll
        for (int i = 0; i < 8; ++i)
          outf[(size_t)(m0 + i) * D_ + n] = (acc[ms][ns][i] + bv) * scale;
      } else {
#pragma unroll
        for (int i = 0; i < 8; ++i)
          outb[(size_t)(m0 + i) * D_ + n] = (bhalf)((acc[ms][ns][i] + bv) * scale);
      }
    }
  }
}

// ---------------------------------------------------------------------------
// Flash attention (transposed): per block = one (b,h), 128 query rows
// (8 waves x 16 q). K (32t x 64d) and Vt (64vd x 32t) tiles are staged into
// LDS by the Tensor Data Mover, double-buffered (wave 0 issues descriptors
// for chunk i+1, waits TENSORcnt<=2 so chunk i is resident, then a workgroup
// barrier releases all 8 waves). TDM row padding (16B) makes the 16
// fragment rows hit 16 distinct bank groups. Wave computes S^T = K*Q^T,
// online softmax over t, then O^T += Vt * P^T; P^T reuses the WMMA D
// registers as the next B operand after one cross-half shuffle exchange.
// ---------------------------------------------------------------------------
#define KSTR 72   // K tile LDS row stride in bf16 (144B = 128B + 16B pad)
#define VSTR 40   // V tile LDS row stride in bf16 (80B = 64B + 16B pad)
#define NIT  (T_ / 32)

__global__ __launch_bounds__(256)
void attn_kernel(const bhalf* __restrict__ Q, const bhalf* __restrict__ K,
                 const bhalf* __restrict__ Vt, bhalf* __restrict__ ctx)
{
  __shared__ bhalf ldsK[2][32 * KSTR];
  __shared__ bhalf ldsV[2][64 * VSTR];

  const int tid  = threadIdx.x;
  const int lane = tid & 31;
  const int wave = tid >> 5;
  const int l16  = lane & 15;
  const int lh   = lane >> 4;
  const int bh = blockIdx.y;
  const int b  = bh >> 4;
  const int h  = bh & (H_ - 1);
  const int q  = blockIdx.x * 128 + wave * 16 + l16;   // this lane's query col

  const bhalf* Kb = K  + ((size_t)b * T_ * H_ + h) * HD_;   // + t*H*64 + d
  const bhalf* Vb = Vt + ((size_t)b * H_ + h) * HD_ * T_;   // + vd*T + t

  // Q as B operand: row n=q, contiguous k(d) in [32*c + 16*lh, +15]
  const bhalf* qrow = Q + ((size_t)(b * T_ + q) * H_ + h) * HD_;
  ABfrag bq[2];
#pragma unroll
  for (int c = 0; c < 2; ++c) {
    int d0 = c * 32 + lh * 16;
    bq[c].h[0] = *(const v8bf*)(qrow + d0);
    bq[c].h[1] = *(const v8bf*)(qrow + d0 + 8);
  }

  v8f o[4];
#pragma unroll
  for (int s = 0; s < 4; ++s) o[s] = (v8f){};
  float rm = -INFINITY, rs = 0.f;

  // prologue: TDM loads for chunk 0 into buffer 0
  if (wave == 0) {
    tdm_load_2d(lds_off(&ldsK[0][0]), Kb, /*tile*/HD_, 32,
                /*tensor*/HD_, T_, H_ * HD_, /*pad int(32dw)*/4, /*amt 4dw*/3, 1);
    tdm_load_2d(lds_off(&ldsV[0][0]), Vb, /*tile*/32, HD_,
                /*tensor*/T_, HD_, T_, /*pad int(16dw)*/3, /*amt 4dw*/3, 1);
  }

  for (int it = 0; it < NIT; ++it) {
    const int tc = it * 32;
    if (wave == 0) {
      if (it + 1 < NIT) {
        const int tn = tc + 32;
        tdm_load_2d(lds_off(&ldsK[(it + 1) & 1][0]),
                    Kb + (size_t)tn * (H_ * HD_), HD_, 32,
                    HD_, T_, H_ * HD_, 4, 3, 1);
        tdm_load_2d(lds_off(&ldsV[(it + 1) & 1][0]),
                    Vb + tn, 32, HD_,
                    T_, HD_, T_, 3, 3, 1);
        __builtin_amdgcn_s_wait_tensorcnt(2);   // chunk `it` resident
      } else {
        __builtin_amdgcn_s_wait_tensorcnt(0);
      }
    }
    __syncthreads();
    const bhalf* lk = &ldsK[it & 1][0];
    const bhalf* lv = &ldsV[it & 1][0];

    // ---- S^T tiles: s0 -> t in [tc, tc+16), s1 -> [tc+16, tc+32), n = q
    v8f s0 = (v8f){}, s1 = (v8f){};
#pragma unroll
    for (int ds = 0; ds < 2; ++ds) {           // d chunks of 32
      int kb = ds * 32 + lh * 16;              // A-frag contiguous k
      ABfrag a0, a1;
      a0.h[0] = *(const v8bf*)&lk[l16 * KSTR + kb];
      a0.h[1] = *(const v8bf*)&lk[l16 * KSTR + kb + 8];
      a1.h[0] = *(const v8bf*)&lk[(16 + l16) * KSTR + kb];
      a1.h[1] = *(const v8bf*)&lk[(16 + l16) * KSTR + kb + 8];
      s0 = wmma_bf16(a0.v, bq[ds].v, s0);
      s1 = wmma_bf16(a1.v, bq[ds].v, s1);
    }

    // ---- online softmax over t for this lane's q (Q pre-scaled by 1/8)
    float mx = -INFINITY;
#pragma unroll
    for (int i = 0; i < 8; ++i) { mx = fmaxf(mx, s0[i]); mx = fmaxf(mx, s1[i]); }
    mx = fmaxf(mx, __shfl_xor(mx, 16, 32));
    float nm    = fmaxf(rm, mx);
    float alpha = __expf(rm - nm);
    float p0[8], p1[8], ls = 0.f;
#pragma unroll
    for (int i = 0; i < 8; ++i) {
      p0[i] = __expf(s0[i] - nm);
      p1[i] = __expf(s1[i] - nm);
      ls += p0[i] + p1[i];
    }
    ls += __shfl_xor(ls, 16, 32);
    rs = rs * alpha + ls;
    rm = nm;
#pragma unroll
    for (int s = 0; s < 4; ++s) o[s] = o[s] * alpha;

    // ---- build P^T as B operand: exchange halves so lane half lh holds
    //      t_local in [16*lh, 16*lh+15] contiguously
    ABfrag pb;
#pragma unroll
    for (int i = 0; i < 8; ++i) {
      float send = lh ? p0[i] : p1[i];
      float recv = __shfl_xor(send, 16, 32);
      float lo = lh ? recv  : p0[i];
      float hi = lh ? p1[i] : recv;
      pb.v[i]     = (bhalf)lo;
      pb.v[i + 8] = (bhalf)hi;
    }

    // ---- O^T += Vt_tile(16 vd x 32 t) * P^T(32 t x 16 q)
#pragma unroll
    for (int sub = 0; sub < 4; ++sub) {
      ABfrag av;
      av.h[0] = *(const v8bf*)&lv[(sub * 16 + l16) * VSTR + lh * 16];
      av.h[1] = *(const v8bf*)&lv[(sub * 16 + l16) * VSTR + lh * 16 + 8];
      o[sub] = wmma_bf16(av.v, pb.v, o[sub]);
    }
    __syncthreads();   // protect buffers before TDM overwrites next parity
  }

  // ---- normalize and store ctx[b][q][h*64 + vd] (8 contiguous vd per lane)
  float inv = 1.0f / rs;
#pragma unroll
  for (int sub = 0; sub < 4; ++sub) {
    v8bf pk;
#pragma unroll
    for (int i = 0; i < 8; ++i) pk[i] = (bhalf)(o[sub][i] * inv);
    *(v8bf*)&ctx[(size_t)(b * T_ + q) * D_ + h * HD_ + sub * 16 + lh * 8] = pk;
  }
}

// ---------------------------------------------------------------------------
extern "C" void kernel_launch(void* const* d_in, const int* in_sizes, int n_in,
                              void* d_out, int out_size, void* d_ws, size_t ws_size,
                              hipStream_t stream) {
  (void)in_sizes; (void)n_in; (void)out_size; (void)ws_size;
  const float* q_in = (const float*)d_in[0];
  const float* v_in = (const float*)d_in[1];
  const float* Wq   = (const float*)d_in[2];
  const float* bq   = (const float*)d_in[3];
  const float* Wk   = (const float*)d_in[4];
  const float* bk   = (const float*)d_in[5];
  const float* Wv   = (const float*)d_in[6];
  const float* bv   = (const float*)d_in[7];
  const float* Wo   = (const float*)d_in[8];
  const float* bo   = (const float*)d_in[9];
  float* out = (float*)d_out;

  char* ws = (char*)d_ws;
  const size_t MB = (size_t)1 << 20;
  bhalf* WqT = (bhalf*)(ws + 0 * MB);    // 2 MB each
  bhalf* WkT = (bhalf*)(ws + 2 * MB);
  bhalf* WvT = (bhalf*)(ws + 4 * MB);
  bhalf* WoT = (bhalf*)(ws + 6 * MB);
  bhalf* Qb  = (bhalf*)(ws + 8 * MB);    // 16 MB each
  bhalf* Kb  = (bhalf*)(ws + 24 * MB);
  bhalf* VtB = (bhalf*)(ws + 40 * MB);
  bhalf* Cx  = (bhalf*)(ws + 56 * MB);   // total 72 MB

  dim3 blk(256);
  // 1) weight transpose+convert
  wconv_kernel<<<dim3((D_ * D_) / 256), blk, 0, stream>>>(Wq, WqT);
  wconv_kernel<<<dim3((D_ * D_) / 256), blk, 0, stream>>>(Wk, WkT);
  wconv_kernel<<<dim3((D_ * D_) / 256), blk, 0, stream>>>(Wv, WvT);
  wconv_kernel<<<dim3((D_ * D_) / 256), blk, 0, stream>>>(Wo, WoT);

  // 2) QKV projections (Q fused with 1/sqrt(64))
  dim3 gproj(D_ / 128, BT_ / 128);
  gemm_kernel<<<gproj, blk, 0, stream>>>(q_in, nullptr, WqT, bq, 0.125f, 0,
                                         Qb, nullptr, nullptr);
  gemm_kernel<<<gproj, blk, 0, stream>>>(v_in, nullptr, WkT, bk, 1.0f, 1,
                                         Kb, nullptr, nullptr);
  gemm_kernel<<<gproj, blk, 0, stream>>>(v_in, nullptr, WvT, bv, 1.0f, 2,
                                         nullptr, VtB, nullptr);

  // 3) flash attention (TDM double-buffered K/V tiles)
  dim3 gattn(T_ / 128, B_ * H_);
  attn_kernel<<<gattn, blk, 0, stream>>>(Qb, Kb, VtB, Cx);

  // 4) output projection -> f32
  gemm_kernel<<<gproj, blk, 0, stream>>>(nullptr, Cx, WoT, bo, 1.0f, 3,
                                         nullptr, nullptr, out);
}
